// GroupedQueryAttention_26113401160302
// MI455X (gfx1250) — compile-verified, matrix operand order checked by reading, not run on hardware
//
#include <hip/hip_runtime.h>
#include <hip/hip_bf16.h>

// ---------------------------------------------------------------------------
// GroupedQueryAttention forward for gfx1250 (MI455X), bf16 WMMA pipeline
// with TDM (tensor_load_to_lds) double-buffered K/V staging in attention.
// B=2, S=2048, HIDDEN=2048, NUM_HEADS=32, KV_HEADS=8, HEAD_DIM=64
// ---------------------------------------------------------------------------

#define B_SZ     2
#define S_LEN    2048
#define HIDDEN   2048
#define NHEADS   32
#define KVHEADS  8
#define HDIM     64
#define LDSP     72   // LDS row stride in ushorts (144B, multiple of 16B)

typedef __attribute__((ext_vector_type(16))) __bf16 bf16x16;
typedef __attribute__((ext_vector_type(8)))  float  f32x8;
typedef __attribute__((ext_vector_type(4)))  unsigned int uint32x4;
typedef __attribute__((ext_vector_type(8)))  int    int32x8;
typedef __attribute__((ext_vector_type(4)))  int    int32x4;

#if defined(__gfx1250__) && __has_builtin(__builtin_amdgcn_tensor_load_to_lds) && \
    __has_builtin(__builtin_amdgcn_s_wait_tensorcnt)
#define USE_TDM 1
#else
#define USE_TDM 0
#endif

union FragU {
    bf16x16 v;
    unsigned short u[16];
    uint4 q[2];
};

__device__ __forceinline__ unsigned short f32_to_bf16(float f) {
    union { float f; unsigned int u; } a; a.f = f;
    unsigned int u = a.u;
    unsigned int r = u + 0x7FFFu + ((u >> 16) & 1u);   // round-to-nearest-even
    return (unsigned short)(r >> 16);
}

// Load a 16(rows) x 32(K) bf16 fragment from a row-major matrix.
// Per-lane layout (ISA 7.12.2, 16-bit A/B 16x32):
//   lanes 0-15 : row = lane,    K = {0..7, 16..23}
//   lanes 16-31: row = lane-16, K = {8..15, 24..31}
// => two contiguous 16-byte loads per lane.
__device__ __forceinline__ bf16x16 load_frag(const unsigned short* base, int ld, int lane) {
    const unsigned short* p = base + (long)(lane & 15) * ld + ((lane >> 4) << 3);
    FragU f;
    f.q[0] = *(const uint4*)(p);
    f.q[1] = *(const uint4*)(p + 16);
    return f.v;
}

// B fragment with strided columns (V tile in LDS): lane n holds output column
// (lane&15); element j is key row kbase+khalf+j / kbase+16+khalf+j.
__device__ __forceinline__ bf16x16 load_fragB_cols(const unsigned short* vbase, int ld,
                                                   int kbase, int lane) {
    int n = lane & 15;
    int khalf = (lane >> 4) << 3;
    FragU f;
#pragma unroll
    for (int j = 0; j < 8; ++j) {
        f.u[j]     = vbase[(kbase + khalf + j) * ld + n];
        f.u[8 + j] = vbase[(kbase + 16 + khalf + j) * ld + n];
    }
    return f.v;
}

__device__ __forceinline__ f32x8 wmma_bf16(bf16x16 a, bf16x16 b, f32x8 c) {
    return __builtin_amdgcn_wmma_f32_16x16x32_bf16(
        false, a, false, b, (short)0, c, false, false);
}

#if USE_TDM
// TDM load of a 64x64 bf16 tile (row stride 512 elems in memory) into LDS,
// with DMA-side padding producing 72-ushort (144B) LDS rows.
// D# layout per CDNA5 ISA.
__device__ __forceinline__ void tdm_load_tile64x64_bf16(unsigned lds_off,
                                                        const unsigned short* gsrc) {
    unsigned long long ga = (unsigned long long)(uintptr_t)gsrc;
    uint32x4 g0;
    g0[0] = 1u;                                   // count=1, user descriptor
    g0[1] = lds_off;                              // lds_addr (bytes)
    g0[2] = (unsigned)ga;                         // global_addr[31:0]
    g0[3] = (unsigned)((ga >> 32) & 0x01FFFFFFu)  // global_addr[56:32]
          | (2u << 30);                           // type=2 ("image")
    int32x8 g1;
    g1[0] = (int)((1u << 16)      // data_size = 2 bytes
                | (1u << 20)      // pad_enable
                | (4u << 22)      // pad_interval: 32 DWORDs = 128B (one row)
                | (3u << 25));    // pad_amount:   4 DWORDs = 16B  -> 144B rows
    g1[1] = (int)(64u << 16);     // tensor_dim0 = 64
    g1[2] = (int)(64u << 16);     // tensor_dim1 = 64
    g1[3] = (int)(64u << 16);     // tile_dim0   = 64
    g1[4] = (int)(64u);           // tile_dim1   = 64, tile_dim2 = 0
    g1[5] = (int)(KVHEADS * HDIM);// tensor_dim0_stride = 512
    g1[6] = 0;
    g1[7] = 0;
    int32x4 z4 = {0, 0, 0, 0};
#if __clang_major__ >= 23
    int32x8 z8 = {0, 0, 0, 0, 0, 0, 0, 0};
    __builtin_amdgcn_tensor_load_to_lds(g0, g1, z4, z4, z8, 0);
#else
    __builtin_amdgcn_tensor_load_to_lds(g0, g1, z4, z4, 0);
#endif
}
#endif // USE_TDM

// ---------------------------------------------------------------------------
// Elementwise f32 -> bf16 conversion
// ---------------------------------------------------------------------------
__global__ void cvt_f32_bf16_k(const float* __restrict__ in,
                               unsigned short* __restrict__ out, long n) {
    long i = (long)blockIdx.x * blockDim.x + threadIdx.x;
    if (i < n) out[i] = f32_to_bf16(in[i]);
}

// ---------------------------------------------------------------------------
// Fused RoPE + f32->bf16 convert for Q/K.  Layout [B*S][nheads*64].
// ---------------------------------------------------------------------------
__global__ void rope_cvt_k(const float* __restrict__ in,
                           unsigned short* __restrict__ out,
                           int nheads) {
    long idx = (long)blockIdx.x * blockDim.x + threadIdx.x;
    long total = (long)B_SZ * S_LEN * nheads * 32;
    if (idx >= total) return;
    int  i   = (int)(idx & 31);
    long t   = idx >> 5;
    int  h   = (int)(t % nheads);
    long row = t / nheads;                 // b*S + s
    int  s   = (int)(row % S_LEN);
    float inv = __powf(10000.0f, -(float)(2 * i) / 64.0f);
    float sn, cs;
    __sincosf((float)s * inv, &sn, &cs);
    long base = row * ((long)nheads * HDIM) + (long)h * HDIM + i;
    float x0 = in[base], x1 = in[base + 32];
    out[base]      = f32_to_bf16(x0 * cs - x1 * sn);
    out[base + 32] = f32_to_bf16(x1 * cs + x0 * sn);
}

// ---------------------------------------------------------------------------
// WMMA GEMM:  C[M,N] (f32, row-major) = A[M,K] (bf16) * B[N,K]^T (bf16)
// 256 threads = 8 waves; WG tile 64(M) x 256(N); wave tile 32 x 64.
// 6 b128-load clause -> 8 back-to-back WMMAs per k-step (software pipelined
// by keeping all fragments in distinct registers).
// ---------------------------------------------------------------------------
__global__ __launch_bounds__(256)
void gemm_bf16_nt(const unsigned short* __restrict__ A,
                  const unsigned short* __restrict__ B,
                  float* __restrict__ C, int M, int N, int K) {
    int lane = threadIdx.x & 31;
    int wave = threadIdx.x >> 5;
    int mrow  = blockIdx.y * 64 + (wave & 1) * 32;
    int ncol0 = blockIdx.x * 256 + (wave >> 1) * 64;

    f32x8 acc[2][4] = {{{}, {}, {}, {}}, {{}, {}, {}, {}}};

    const unsigned short* Ap0 = A + (long)mrow * K;
    const unsigned short* Ap1 = A + (long)(mrow + 16) * K;
    const unsigned short* Bp0 = B + (long)(ncol0)      * K;
    const unsigned short* Bp1 = B + (long)(ncol0 + 16) * K;
    const unsigned short* Bp2 = B + (long)(ncol0 + 32) * K;
    const unsigned short* Bp3 = B + (long)(ncol0 + 48) * K;

    for (int k0 = 0; k0 < K; k0 += 32) {
        bf16x16 a0 = load_frag(Ap0 + k0, K, lane);
        bf16x16 a1 = load_frag(Ap1 + k0, K, lane);
        bf16x16 b0 = load_frag(Bp0 + k0, K, lane);
        bf16x16 b1 = load_frag(Bp1 + k0, K, lane);
        bf16x16 b2 = load_frag(Bp2 + k0, K, lane);
        bf16x16 b3 = load_frag(Bp3 + k0, K, lane);
        acc[0][0] = wmma_bf16(a0, b0, acc[0][0]);
        acc[0][1] = wmma_bf16(a0, b1, acc[0][1]);
        acc[0][2] = wmma_bf16(a0, b2, acc[0][2]);
        acc[0][3] = wmma_bf16(a0, b3, acc[0][3]);
        acc[1][0] = wmma_bf16(a1, b0, acc[1][0]);
        acc[1][1] = wmma_bf16(a1, b1, acc[1][1]);
        acc[1][2] = wmma_bf16(a1, b2, acc[1][2]);
        acc[1][3] = wmma_bf16(a1, b3, acc[1][3]);
    }

    int rb = (lane >> 4) << 3;
    int cc = lane & 15;
#pragma unroll
    for (int mi = 0; mi < 2; ++mi)
#pragma unroll
        for (int t = 0; t < 4; ++t)
#pragma unroll
            for (int v = 0; v < 8; ++v)
                C[(long)(mrow + mi * 16 + rb + v) * N + ncol0 + t * 16 + cc] =
                    acc[mi][t][v];
}

// ---------------------------------------------------------------------------
// Flash-style causal GQA attention.
// Grid: B*NHEADS*(S/64) WGs, 128 threads (4 waves). Each wave: 16 q rows.
// K/V 64x64 tiles double-buffered in LDS; staged by the Tensor Data Mover
// (wave 0 issues tensor_load_to_lds, tracked with TENSORcnt) when available.
// ---------------------------------------------------------------------------
__global__ __launch_bounds__(128)
void attn_fwd_k(const unsigned short* __restrict__ qh,
                const unsigned short* __restrict__ kh,
                const unsigned short* __restrict__ vh,
                unsigned short* __restrict__ oh) {
    __shared__ unsigned short Kls[2][64 * LDSP];
    __shared__ unsigned short Vls[2][64 * LDSP];
    __shared__ unsigned short Pls[4 * 16 * LDSP];

    int wg  = blockIdx.x;
    int qb  = wg & 31;              // q 64-row tile index
    int h   = (wg >> 5) & (NHEADS - 1);
    int b   = wg >> 10;
    int kvh = h >> 2;               // 4:1 GQA
    int lane = threadIdx.x & 31;
    int wave = threadIdx.x >> 5;

    long rowbase = (long)b * S_LEN;
    int  qrow0   = qb * 64 + wave * 16;

    const unsigned short* qbase =
        qh + (rowbase + qrow0) * (long)(NHEADS * HDIM) + h * HDIM;
    bf16x16 qa0 = load_frag(qbase, NHEADS * HDIM, lane);
    bf16x16 qa1 = load_frag(qbase + 32, NHEADS * HDIM, lane);

    f32x8 acc[4] = {{}, {}, {}, {}};
    float mrow[8], lrow[8];
#pragma unroll
    for (int v = 0; v < 8; ++v) { mrow[v] = -3.0e38f; lrow[v] = 0.0f; }

    int hi8 = (lane >> 4) << 3;
    int nkb = qb + 1;               // causal: key blocks 0..qb

    auto issue_tile = [&](int buf, int kb2) {
        long src = (rowbase + (long)kb2 * 64) * (KVHEADS * HDIM) + kvh * HDIM;
#if USE_TDM
        if (wave == 0) {
            tdm_load_tile64x64_bf16((unsigned)(uintptr_t)&Kls[buf][0], kh + src);
            tdm_load_tile64x64_bf16((unsigned)(uintptr_t)&Vls[buf][0], vh + src);
        }
#else
        for (int t = threadIdx.x; t < 64 * 8; t += 128) {
            int r  = t >> 3;
            int cq = t & 7;
            long s2 = src + (long)r * (KVHEADS * HDIM) + cq * 8;
            *(uint4*)&Kls[buf][r * LDSP + cq * 8] = *(const uint4*)&kh[s2];
            *(uint4*)&Vls[buf][r * LDSP + cq * 8] = *(const uint4*)&vh[s2];
        }
#endif
    };

    issue_tile(0, 0);

    for (int kb = 0; kb < nkb; ++kb) {
        int cur = kb & 1;
        bool more = (kb + 1) < nkb;
        if (more) issue_tile(cur ^ 1, kb + 1);
#if USE_TDM
        if (wave == 0) {
            if (more) __builtin_amdgcn_s_wait_tensorcnt((short)2);
            else      __builtin_amdgcn_s_wait_tensorcnt((short)0);
        }
#endif
        __syncthreads();

        const unsigned short* Kc = &Kls[cur][0];
        const unsigned short* Vc = &Vls[cur][0];
        int krow0 = kb * 64;

        // ---- scores = q @ k^T, 16 x 64 per wave ----
        f32x8 sc[4];
#pragma unroll
        for (int cg = 0; cg < 4; ++cg) {
            bf16x16 kb0 = load_frag(Kc + (cg * 16) * LDSP, LDSP, lane);
            bf16x16 kb1 = load_frag(Kc + (cg * 16) * LDSP + 32, LDSP, lane);
            f32x8 c = {};
            c = wmma_bf16(qa0, kb0, c);
            c = wmma_bf16(qa1, kb1, c);
            sc[cg] = c;
        }

        // ---- scale + causal mask ----
#pragma unroll
        for (int cg = 0; cg < 4; ++cg) {
            int colg = krow0 + cg * 16 + (lane & 15);
#pragma unroll
            for (int v = 0; v < 8; ++v) {
                int rowg = qrow0 + hi8 + v;
                float s  = sc[cg][v] * 0.125f;     // 1/sqrt(64)
                sc[cg][v] = (colg <= rowg) ? s : -3.0e38f;
            }
        }

        // ---- online softmax: block row-max across 16-lane halves ----
        float bmax[8];
#pragma unroll
        for (int v = 0; v < 8; ++v) {
            float m = fmaxf(fmaxf(sc[0][v], sc[1][v]), fmaxf(sc[2][v], sc[3][v]));
            bmax[v] = m;
        }
#pragma unroll
        for (int off = 8; off >= 1; off >>= 1)
#pragma unroll
            for (int v = 0; v < 8; ++v)
                bmax[v] = fmaxf(bmax[v], __shfl_xor(bmax[v], off, 16));

        float scl[8];
#pragma unroll
        for (int v = 0; v < 8; ++v) {
            float mn = fmaxf(mrow[v], bmax[v]);
            scl[v] = __expf(mrow[v] - mn);
            mrow[v] = mn;
        }

        float bsum[8];
#pragma unroll
        for (int v = 0; v < 8; ++v) bsum[v] = 0.0f;
#pragma unroll
        for (int cg = 0; cg < 4; ++cg)
#pragma unroll
            for (int v = 0; v < 8; ++v) {
                float p = __expf(sc[cg][v] - mrow[v]);
                sc[cg][v] = p;
                bsum[v] += p;
            }
#pragma unroll
        for (int off = 8; off >= 1; off >>= 1)
#pragma unroll
            for (int v = 0; v < 8; ++v)
                bsum[v] += __shfl_xor(bsum[v], off, 16);
#pragma unroll
        for (int v = 0; v < 8; ++v)
            lrow[v] = lrow[v] * scl[v] + bsum[v];

#pragma unroll
        for (int t = 0; t < 4; ++t)
#pragma unroll
            for (int v = 0; v < 8; ++v)
                acc[t][v] *= scl[v];

        // ---- stage P (16 x 64 bf16) into this wave's LDS region ----
        unsigned short* P = &Pls[wave * 16 * LDSP];
#pragma unroll
        for (int cg = 0; cg < 4; ++cg)
#pragma unroll
            for (int v = 0; v < 8; ++v)
                P[(hi8 + v) * LDSP + cg * 16 + (lane & 15)] = f32_to_bf16(sc[cg][v]);
        // same-wave LDS ops are in-order; only this wave reads its P region.

        // ---- out += P @ V ----
        bf16x16 pa0 = load_frag(P, LDSP, lane);         // keys 0..31
        bf16x16 pa1 = load_frag(P + 32, LDSP, lane);    // keys 32..63
#pragma unroll
        for (int ct = 0; ct < 4; ++ct) {
            bf16x16 vb0 = load_fragB_cols(Vc + ct * 16, LDSP, 0, lane);
            bf16x16 vb1 = load_fragB_cols(Vc + ct * 16, LDSP, 32, lane);
            acc[ct] = wmma_bf16(pa0, vb0, acc[ct]);
            acc[ct] = wmma_bf16(pa1, vb1, acc[ct]);
        }
        __syncthreads();
    }

    // ---- normalize and write out (bf16) ----
#pragma unroll
    for (int ct = 0; ct < 4; ++ct)
#pragma unroll
        for (int v = 0; v < 8; ++v) {
            long row = rowbase + qrow0 + hi8 + v;
            oh[row * (long)HIDDEN + h * HDIM + ct * 16 + (lane & 15)] =
                f32_to_bf16(acc[ct][v] / lrow[v]);
        }
}

// ---------------------------------------------------------------------------
// Host launcher
// ---------------------------------------------------------------------------
extern "C" void kernel_launch(void* const* d_in, const int* in_sizes, int n_in,
                              void* d_out, int out_size, void* d_ws, size_t ws_size,
                              hipStream_t stream) {
    const float* x  = (const float*)d_in[0];
    const float* Wq = (const float*)d_in[1];
    const float* Wk = (const float*)d_in[2];
    const float* Wv = (const float*)d_in[3];
    const float* Wo = (const float*)d_in[4];
    float* out = (float*)d_out;

    const long BS = (long)B_SZ * S_LEN;   // 4096
    const long H  = HIDDEN;               // 2048
    const long KV = (long)KVHEADS * HDIM; // 512

    char* w = (char*)d_ws;
    auto alloc = [&](size_t bytes) -> char* {
        char* p = w;
        w += (bytes + 255) & ~(size_t)255;
        return p;
    };
    unsigned short* xh  = (unsigned short*)alloc(BS * H * 2);
    unsigned short* wqh = (unsigned short*)alloc(H * H * 2);
    unsigned short* wkh = (unsigned short*)alloc(KV * H * 2);
    unsigned short* wvh = (unsigned short*)alloc(KV * H * 2);
    unsigned short* woh = (unsigned short*)alloc(H * H * 2);
    float*          kf  = (float*)alloc(BS * KV * 4);
    float*          vf  = (float*)alloc(BS * KV * 4);
    unsigned short* qh2 = (unsigned short*)alloc(BS * H * 2);
    unsigned short* kh2 = (unsigned short*)alloc(BS * KV * 2);
    unsigned short* vh2 = (unsigned short*)alloc(BS * KV * 2);
    unsigned short* ah  = (unsigned short*)alloc(BS * H * 2);
    float* qf = out;   // reuse d_out (4096x2048 f32) as Q-proj scratch

    auto blks = [](long n) { return (unsigned)((n + 255) / 256); };

    // 1) bf16 conversions
    cvt_f32_bf16_k<<<blks(BS * H), 256, 0, stream>>>(x,  xh,  BS * H);
    cvt_f32_bf16_k<<<blks(H * H),  256, 0, stream>>>(Wq, wqh, H * H);
    cvt_f32_bf16_k<<<blks(KV * H), 256, 0, stream>>>(Wk, wkh, KV * H);
    cvt_f32_bf16_k<<<blks(KV * H), 256, 0, stream>>>(Wv, wvh, KV * H);
    cvt_f32_bf16_k<<<blks(H * H),  256, 0, stream>>>(Wo, woh, H * H);

    // 2) QKV projections (WMMA GEMM, C = A * B^T), WG tile 64x256
    gemm_bf16_nt<<<dim3(H / 256,  BS / 64), 256, 0, stream>>>(xh, wqh, qf, BS, H,  H);
    gemm_bf16_nt<<<dim3(KV / 256, BS / 64), 256, 0, stream>>>(xh, wkh, kf, BS, KV, H);
    gemm_bf16_nt<<<dim3(KV / 256, BS / 64), 256, 0, stream>>>(xh, wvh, vf, BS, KV, H);

    // 3) RoPE + bf16 convert for Q, K; plain convert for V
    rope_cvt_k<<<blks(BS * NHEADS * 32),  256, 0, stream>>>(qf, qh2, NHEADS);
    rope_cvt_k<<<blks(BS * KVHEADS * 32), 256, 0, stream>>>(kf, kh2, KVHEADS);
    cvt_f32_bf16_k<<<blks(BS * KV), 256, 0, stream>>>(vf, vh2, BS * KV);

    // 4) causal GQA flash attention (TDM-staged K/V tiles)
    attn_fwd_k<<<B_SZ * NHEADS * (S_LEN / 64), 128, 0, stream>>>(qh2, kh2, vh2, ah);

    // 5) output projection (overwrites d_out fully)
    gemm_bf16_nt<<<dim3(H / 256, BS / 64), 256, 0, stream>>>(ah, woh, out, BS, H, H);
}
